// DeformableLSTM_76759655514945
// MI455X (gfx1250) — matrix operand order repaired and checked by reading, not compiled, not run on hardware
//
#include <hip/hip_runtime.h>
#include <hip/hip_bf16.h>
#include <math.h>

// ---------------- problem constants ----------------
#define D_MODEL 512
#define N_HEADS 8
#define N_LEVELS 4
#define N_POINTS 4
#define D_HEAD 64
#define HID 512
#define B_ 16
#define Q_ 64
#define BQ (B_ * Q_)           // 1024
#define T_TOTAL 3840
#define LP 16                  // N_LEVELS * N_POINTS
#define M_VALUE (B_ * T_TOTAL) // 61440
#define M_SCORE (BQ * N_HEADS * LP) // 131072

typedef _Float16 half_t;
typedef half_t v16h __attribute__((ext_vector_type(16)));
typedef float  v8f  __attribute__((ext_vector_type(8)));

// =====================================================================
// Generic split-precision (f16 hi+lo) WMMA GEMM:
//   C[M,N] = (A[M,K] @ B[K,N or N,K if transB] (+ C if accumulate) + bias[n]) * rowscale[m]
// One 16x16 C tile per wave. All dims multiples of 16; K multiple of 32.
// Fragment layouts per CDNA5 ISA 7.12.2.
// =====================================================================
__global__ void wmma_gemm_kernel(const float* __restrict__ A,
                                 const float* __restrict__ Bm,
                                 const float* __restrict__ bias,
                                 const float* __restrict__ rowscale,
                                 float* __restrict__ C,
                                 int M, int N, int K, int transB, int accumulate)
{
    const int lane = threadIdx.x & 31;
    const int wave = threadIdx.x >> 5;
    const int tilesN = N >> 4;
    const int numTiles = (M >> 4) * tilesN;
    const int tile = blockIdx.x * (blockDim.x >> 5) + wave;
    if (tile >= numTiles) return;                 // wave-uniform: EXEC stays all-1s
    const int tileM = tile / tilesN;
    const int tileN = tile - tileM * tilesN;

    const int lan = lane & 15;
    const int hgrp = lane >> 4;                   // 0 or 1 (lane half)
    const int m = tileM * 16 + lan;               // A row for this lane
    const int n = tileN * 16 + lan;               // B/C column for this lane
    const float* __restrict__ arow = A + (size_t)m * K;

    v8f acc = {};
    const int rowBase = tileM * 16 + hgrp * 8;    // C/D: VGPR r <-> M = rowBase + r
    if (accumulate) {
        #pragma unroll
        for (int r = 0; r < 8; ++r)
            acc[r] = C[(size_t)(rowBase + r) * N + n];
    }

    for (int k0 = 0; k0 < K; k0 += 32) {
        if (k0 + 64 < K) __builtin_prefetch(arow + k0 + 64, 0, 1); // global_prefetch_b8

        // ---- A fragment: lane half g covers K = k0+8g..k0+8g+7 and k0+16+8g..+7
        const int kbA = k0 + hgrp * 8;
        v16h a_hi, a_lo;
        #pragma unroll
        for (int j = 0; j < 8; ++j) {
            float x0 = arow[kbA + j];
            half_t h0 = (half_t)x0;
            a_hi[j] = h0; a_lo[j] = (half_t)(x0 - (float)h0);
            float x1 = arow[kbA + 16 + j];
            half_t h1 = (half_t)x1;
            a_hi[8 + j] = h1; a_lo[8 + j] = (half_t)(x1 - (float)h1);
        }

        // ---- B fragment: lane half g covers K = k0+8g..k0+8g+7 and k0+16+8g..+7
        const int kbB = k0 + hgrp * 8;
        v16h b_hi, b_lo;
        #pragma unroll
        for (int e = 0; e < 16; ++e) {
            const int kk = kbB + (e < 8 ? e : e + 8);
            float x = transB ? Bm[(size_t)n * K + kk] : Bm[(size_t)kk * N + n];
            half_t h = (half_t)x;
            b_hi[e] = h; b_lo[e] = (half_t)(x - (float)h);
        }

        // split-precision product: hi*hi + hi*lo + lo*hi (~fp32 accuracy, f32 accum)
        acc = __builtin_amdgcn_wmma_f32_16x16x32_f16(false, a_hi, false, b_hi, (short)0, acc, false, false);
        acc = __builtin_amdgcn_wmma_f32_16x16x32_f16(false, a_hi, false, b_lo, (short)0, acc, false, false);
        acc = __builtin_amdgcn_wmma_f32_16x16x32_f16(false, a_lo, false, b_hi, (short)0, acc, false, false);
    }

    const float bcol = bias ? bias[n] : 0.0f;
    #pragma unroll
    for (int r = 0; r < 8; ++r) {
        const int row = rowBase + r;
        float v = acc[r] + bcol;
        if (rowscale) v *= rowscale[row];
        C[(size_t)row * N + n] = v;
    }
}

// =====================================================================
// Fused additive-attention scores:
//   scores[m] = sum_n tanh( (clip@W_ctx)[m,n] + b_ctx[n] + attn_h[m>>7, n] ) * W_alpha[n] + b_alpha
// One wave per 16-row group (one (bq,h) pair's 16 samples). WMMA over K=64,
// W_alpha reduction folded into epilogue -> no 256 MB 'dot' intermediate.
// =====================================================================
__global__ void scores_kernel(const float* __restrict__ clip,
                              const float* __restrict__ Wctx,
                              const float* __restrict__ bctx,
                              const float* __restrict__ attnh,
                              const float* __restrict__ Walpha,
                              const float* __restrict__ balpha,
                              float* __restrict__ scores)
{
    const int lane = threadIdx.x & 31;
    const int wave = threadIdx.x >> 5;
    const int group = blockIdx.x * (blockDim.x >> 5) + wave;   // 8192 groups
    if (group >= (M_SCORE >> 4)) return;
    const int mBase = group * 16;
    const int lan = lane & 15;
    const int hgrp = lane >> 4;

    // A fragments (K=64 -> two 32-wide slices), invariant across n-tiles
    const float* __restrict__ arow = clip + (size_t)(mBase + lan) * D_HEAD;
    v16h ah[2], al[2];
    #pragma unroll
    for (int kk = 0; kk < 2; ++kk) {
        const int kbA = kk * 32 + hgrp * 8;
        #pragma unroll
        for (int j = 0; j < 8; ++j) {
            float x0 = arow[kbA + j];
            half_t h0 = (half_t)x0;
            ah[kk][j] = h0; al[kk][j] = (half_t)(x0 - (float)h0);
            float x1 = arow[kbA + 16 + j];
            half_t h1 = (half_t)x1;
            ah[kk][8 + j] = h1; al[kk][8 + j] = (half_t)(x1 - (float)h1);
        }
    }

    const int rowBase = mBase + hgrp * 8;
    const int bq = mBase >> 7;            // 128 rows per (b,q); group never straddles
    float rsum[8];
    #pragma unroll
    for (int r = 0; r < 8; ++r) rsum[r] = 0.0f;

    for (int nt = 0; nt < (HID >> 4); ++nt) {
        const int n = nt * 16 + lan;
        v8f acc = {};
        #pragma unroll
        for (int kk = 0; kk < 2; ++kk) {
            const int kbB = kk * 32 + hgrp * 8;
            v16h b_hi, b_lo;
            #pragma unroll
            for (int e = 0; e < 16; ++e) {
                float x = Wctx[(size_t)(kbB + (e < 8 ? e : e + 8)) * HID + n];
                half_t h = (half_t)x;
                b_hi[e] = h; b_lo[e] = (half_t)(x - (float)h);
            }
            acc = __builtin_amdgcn_wmma_f32_16x16x32_f16(false, ah[kk], false, b_hi, (short)0, acc, false, false);
            acc = __builtin_amdgcn_wmma_f32_16x16x32_f16(false, ah[kk], false, b_lo, (short)0, acc, false, false);
            acc = __builtin_amdgcn_wmma_f32_16x16x32_f16(false, al[kk], false, b_hi, (short)0, acc, false, false);
        }
        const float bc = bctx[n];
        const float ahn = attnh[(size_t)bq * HID + n];
        const float wa = Walpha[n];
        #pragma unroll
        for (int r = 0; r < 8; ++r)
            rsum[r] += tanhf(acc[r] + bc + ahn) * wa;
    }

    // reduce the 16 column-lanes of each half-wave
    const float ba = balpha[0];
    #pragma unroll
    for (int r = 0; r < 8; ++r) {
        float s = rsum[r];
        s += __shfl_xor(s, 1, 16);
        s += __shfl_xor(s, 2, 16);
        s += __shfl_xor(s, 4, 16);
        s += __shfl_xor(s, 8, 16);
        if (lan == 0) scores[rowBase + r] = s + ba;
    }
}

// =====================================================================
// Element-wise / gather kernels
// =====================================================================
__global__ void build_hs_kernel(const float* __restrict__ h0,
                                const float* __restrict__ query,
                                float* __restrict__ hs)
{
    int idx = blockIdx.x * blockDim.x + threadIdx.x;
    if (idx >= BQ * 2 * D_MODEL) return;
    int j = idx & 1023;
    int m = idx >> 10;
    hs[idx] = (j < D_MODEL) ? h0[(size_t)m * D_MODEL + j]
                            : query[(size_t)m * D_MODEL + (j - D_MODEL)];
}

__global__ void softmax16_kernel(const float* __restrict__ x,
                                 float* __restrict__ y, int ngroups)
{
    int g = blockIdx.x * blockDim.x + threadIdx.x;
    if (g >= ngroups) return;
    const float* in = x + (size_t)g * 16;
    float mx = in[0];
    #pragma unroll
    for (int i = 1; i < 16; ++i) mx = fmaxf(mx, in[i]);
    float s = 0.0f, e[16];
    #pragma unroll
    for (int i = 0; i < 16; ++i) { e[i] = expf(in[i] - mx); s += e[i]; }
    float inv = 1.0f / s;
    float* out = y + (size_t)g * 16;
    #pragma unroll
    for (int i = 0; i < 16; ++i) out[i] = e[i] * inv;
}

// 1-D deformable sampling: linear interp + OOB-zero + attention weight
__global__ void sample_kernel(const float* __restrict__ value,   // [B,T_TOTAL,H,Dh]
                              const float* __restrict__ refp,    // [B,Q,4,1]
                              const float* __restrict__ offb,    // [BQ,128]
                              const float* __restrict__ aw,      // [BQ*H,16]
                              const int* __restrict__ shapes,
                              const int* __restrict__ starts,
                              float* __restrict__ clip)          // [BQ*H*16,64]
{
    int idx = blockIdx.x * blockDim.x + threadIdx.x;
    if (idx >= BQ * N_HEADS * LP * D_HEAD) return;
    const int d  = idx & 63;
    const int lp = (idx >> 6) & 15;
    const int h  = (idx >> 10) & 7;
    const int bq = idx >> 13;
    const int b  = bq >> 6;
    const int l  = lp >> 2;
    const int T = shapes[l];
    const int s = starts[l];

    const float ref = refp[(size_t)bq * N_LEVELS + l];
    const float off = offb[(size_t)bq * (N_HEADS * LP) + h * LP + lp];
    // x = loc*T - 0.5 with loc = ref + off/T  ->  x = ref*T + off - 0.5
    const float x  = ref * (float)T + off - 0.5f;
    const float x0 = floorf(x);
    const float w  = x - x0;
    const int i0 = (int)x0;
    const int i1 = i0 + 1;

    float g0 = 0.0f, g1 = 0.0f;
    const size_t vb = ((size_t)b * T_TOTAL + s) * D_MODEL + h * D_HEAD + d;
    if (i0 >= 0 && i0 < T) g0 = value[vb + (size_t)i0 * D_MODEL];
    if (i1 >= 0 && i1 < T) g1 = value[vb + (size_t)i1 * D_MODEL];

    const float awv = aw[((size_t)bq * N_HEADS + h) * LP + lp];
    clip[((size_t)((bq * N_HEADS + h) * LP + lp)) * D_HEAD + d] =
        (g0 * (1.0f - w) + g1 * w) * awv;
}

__global__ void attnres_kernel(const float* __restrict__ wt,     // [BQ*H,16]
                               const float* __restrict__ clip,   // [BQ*H*16,64]
                               float* __restrict__ attnres)      // [BQ,512]
{
    int idx = blockIdx.x * blockDim.x + threadIdx.x;
    if (idx >= BQ * D_MODEL) return;
    const int d  = idx & 63;
    const int h  = (idx >> 6) & 7;
    const int bq = idx >> 9;
    const size_t base = (size_t)(bq * N_HEADS + h) * LP;
    float acc = 0.0f;
    #pragma unroll
    for (int p = 0; p < LP; ++p)
        acc += wt[base + p] * clip[(base + p) * D_HEAD + d];
    attnres[(size_t)bq * D_MODEL + h * D_HEAD + d] = acc;
}

__global__ void build_xin_kernel(const float* __restrict__ token,
                                 const float* __restrict__ attnres,
                                 const float* __restrict__ query,
                                 float* __restrict__ xin)
{
    int idx = blockIdx.x * blockDim.x + threadIdx.x;
    if (idx >= BQ * 3 * D_MODEL) return;
    int j = idx % (3 * D_MODEL);
    int m = idx / (3 * D_MODEL);
    float v;
    if (j < D_MODEL)          v = token[(size_t)m * D_MODEL + j];
    else if (j < 2 * D_MODEL) v = attnres[(size_t)m * D_MODEL + (j - D_MODEL)];
    else                      v = query[(size_t)m * D_MODEL + (j - 2 * D_MODEL)];
    xin[idx] = v;
}

__device__ __forceinline__ float sigmoidf_(float x) { return 1.0f / (1.0f + expf(-x)); }

__global__ void lstm_kernel(const float* __restrict__ gates,  // [BQ,2048] = [i|f|g|o]
                            const float* __restrict__ c0,
                            float* __restrict__ out)
{
    int idx = blockIdx.x * blockDim.x + threadIdx.x;
    if (idx >= BQ * D_MODEL) return;
    const int n = idx & 511;
    const int m = idx >> 9;
    const float* g = gates + (size_t)m * (4 * D_MODEL);
    const float gi = g[n];
    const float gf = g[D_MODEL + n];
    const float gg = g[2 * D_MODEL + n];
    const float go = g[3 * D_MODEL + n];
    const float c1 = sigmoidf_(gf) * c0[idx] + sigmoidf_(gi) * tanhf(gg);
    out[idx] = sigmoidf_(go) * tanhf(c1);
}

// =====================================================================
// host-side launch
// =====================================================================
extern "C" void kernel_launch(void* const* d_in, const int* in_sizes, int n_in,
                              void* d_out, int out_size, void* d_ws, size_t ws_size,
                              hipStream_t stream) {
    (void)in_sizes; (void)n_in; (void)out_size; (void)ws_size;
    const float* token  = (const float*)d_in[0];
    const float* h0     = (const float*)d_in[1];
    const float* c0     = (const float*)d_in[2];
    const float* query  = (const float*)d_in[3];
    const float* refp   = (const float*)d_in[4];
    const int*   tshape = (const int*)d_in[5];
    const int*   lstart = (const int*)d_in[6];
    const float* enc    = (const float*)d_in[7];
    const float* mask   = (const float*)d_in[8];
    const float* Wv     = (const float*)d_in[9];
    const float* bv     = (const float*)d_in[10];
    const float* Woff   = (const float*)d_in[11];
    const float* boff   = (const float*)d_in[12];
    const float* Waw    = (const float*)d_in[13];
    const float* baw    = (const float*)d_in[14];
    const float* Wctx   = (const float*)d_in[15];
    const float* bctx   = (const float*)d_in[16];
    const float* Whs    = (const float*)d_in[17];
    const float* bhs    = (const float*)d_in[18];
    const float* Walpha = (const float*)d_in[19];
    const float* balpha = (const float*)d_in[20];
    const float* Wih    = (const float*)d_in[21];
    const float* Whh    = (const float*)d_in[22];
    float* out = (float*)d_out;

    // workspace layout (floats); x_in/gates alias value (dead after sampling)
    float* ws = (float*)d_ws;
    float* v_value   = ws;                                  // 61440*512
    float* v_hs      = v_value + (size_t)M_VALUE * D_MODEL; // 1024*1024
    float* v_off     = v_hs + (size_t)BQ * 2 * D_MODEL;     // 1024*128
    float* v_awlog   = v_off + (size_t)BQ * 128;            // 1024*128
    float* v_aw      = v_awlog + (size_t)BQ * 128;          // 1024*128
    float* v_clip    = v_aw + (size_t)BQ * 128;             // 131072*64
    float* v_attnh   = v_clip + (size_t)M_SCORE * D_HEAD;   // 1024*512
    float* v_scores  = v_attnh + (size_t)BQ * HID;          // 131072
    float* v_wt      = v_scores + (size_t)M_SCORE;          // 131072
    float* v_attnres = v_wt + (size_t)M_SCORE;              // 1024*512
    float* v_xin     = v_value;                             // 1024*1536 (alias)
    float* v_gates   = v_value + (size_t)BQ * 3 * D_MODEL;  // 1024*2048 (alias)

    const int TPB = 256;           // 8 wave32 waves
    const int WPB = TPB / 32;

    // 1. hs = [h_last | query]
    build_hs_kernel<<<(BQ * 2 * D_MODEL + TPB - 1) / TPB, TPB, 0, stream>>>(h0, query, v_hs);

    // 2. value = (E @ W_value + b_value) * mask   (M=61440,N=512,K=512)
    {
        int tiles = (M_VALUE / 16) * (D_MODEL / 16);
        wmma_gemm_kernel<<<(tiles + WPB - 1) / WPB, TPB, 0, stream>>>(
            enc, Wv, bv, mask, v_value, M_VALUE, D_MODEL, D_MODEL, 0, 0);
    }

    // 3. off = hs @ W_off + b_off   (1024x1024x128)
    {
        int tiles = (BQ / 16) * (128 / 16);
        wmma_gemm_kernel<<<(tiles + WPB - 1) / WPB, TPB, 0, stream>>>(
            v_hs, Woff, boff, nullptr, v_off, BQ, 128, 2 * D_MODEL, 0, 0);
        wmma_gemm_kernel<<<(tiles + WPB - 1) / WPB, TPB, 0, stream>>>(
            v_hs, Waw, baw, nullptr, v_awlog, BQ, 128, 2 * D_MODEL, 0, 0);
    }

    // 4. aw softmax over the 16 (level,point) slots per (bq, head)
    softmax16_kernel<<<(BQ * N_HEADS + TPB - 1) / TPB, TPB, 0, stream>>>(
        v_awlog, v_aw, BQ * N_HEADS);

    // 5. deformable sampling -> clip_feats (value fits in 192 MB L2)
    sample_kernel<<<(BQ * N_HEADS * LP * D_HEAD + TPB - 1) / TPB, TPB, 0, stream>>>(
        v_value, refp, v_off, v_aw, tshape, lstart, v_clip);

    // 6. attn_h = h_last @ W_hs + b_hs
    {
        int tiles = (BQ / 16) * (HID / 16);
        wmma_gemm_kernel<<<(tiles + WPB - 1) / WPB, TPB, 0, stream>>>(
            h0, Whs, bhs, nullptr, v_attnh, BQ, HID, D_MODEL, 0, 0);
    }

    // 7. fused additive-attention scores (WMMA + tanh + W_alpha reduce)
    scores_kernel<<<((M_SCORE / 16) + WPB - 1) / WPB, TPB, 0, stream>>>(
        v_clip, Wctx, bctx, v_attnh, Walpha, balpha, v_scores);

    // 8. softmax over samples, then weighted sum -> attn_res
    softmax16_kernel<<<(BQ * N_HEADS + TPB - 1) / TPB, TPB, 0, stream>>>(
        v_scores, v_wt, BQ * N_HEADS);
    attnres_kernel<<<(BQ * D_MODEL + TPB - 1) / TPB, TPB, 0, stream>>>(
        v_wt, v_clip, v_attnres);

    // 9. x_in = [token | attn_res | query]; gates = x_in@W_ih.T + h_last@W_hh.T
    build_xin_kernel<<<(BQ * 3 * D_MODEL + TPB - 1) / TPB, TPB, 0, stream>>>(
        token, v_attnres, query, v_xin);
    {
        int tiles = (BQ / 16) * ((4 * D_MODEL) / 16);
        wmma_gemm_kernel<<<(tiles + WPB - 1) / WPB, TPB, 0, stream>>>(
            v_xin, Wih, nullptr, nullptr, v_gates, BQ, 4 * D_MODEL, 3 * D_MODEL, 1, 0);
        wmma_gemm_kernel<<<(tiles + WPB - 1) / WPB, TPB, 0, stream>>>(
            h0, Whh, nullptr, nullptr, v_gates, BQ, 4 * D_MODEL, D_MODEL, 1, 1);
    }

    // 10. LSTM cell pointwise
    lstm_kernel<<<(BQ * D_MODEL + TPB - 1) / TPB, TPB, 0, stream>>>(v_gates, c0, out);
}
